// Transcoder_84954453115205
// MI455X (gfx1250) — compile-verified, matrix operand order checked by reading, not run on hardware
//
#include <hip/hip_runtime.h>
#include <stdint.h>

#define TOKENS 4096
#define D_IN   2048
#define N_FEAT 32768
#define D_OUT  2048
#define TOPK   64

typedef __attribute__((ext_vector_type(16))) __bf16 v16bf;
typedef __attribute__((ext_vector_type(8)))  float  v8f;

// ---------------- f32 -> bf16 (round-to-nearest-even) ----------------
__device__ __forceinline__ unsigned short f2bf(float f) {
  unsigned int u = __float_as_uint(f);
  unsigned int r = u + 0x7FFFu + ((u >> 16) & 1u);
  return (unsigned short)(r >> 16);
}

__global__ void cvt_f32_bf16(const float* __restrict__ in,
                             unsigned short* __restrict__ out, int n4) {
  int i = blockIdx.x * blockDim.x + threadIdx.x;
  int stride = gridDim.x * blockDim.x;
  for (; i < n4; i += stride) {
    float4 f = reinterpret_cast<const float4*>(in)[i];
    ushort4 b;
    b.x = f2bf(f.x); b.y = f2bf(f.y); b.z = f2bf(f.z); b.w = f2bf(f.w);
    reinterpret_cast<ushort4*>(out)[i] = b;
  }
}

// ---------------- WMMA GEMM: C[M,N] = A[M,K] @ B[N,K]^T (+bias, relu) ----
// A, B bf16 row-major (K contiguous). 256 threads = 8 waves.
// Block tile 128(M) x 128(N) x 32(K). Wave tile 32(M) x 64(N) = 2x4 WMMA.
// Double-buffered LDS filled with GLOBAL_LOAD_ASYNC_TO_LDS_B128 (ASYNCcnt),
// one workgroup barrier per K-step; stage i+1 copy overlaps stage i WMMAs.
__global__ __launch_bounds__(256)
void wmma_gemm_bt(const unsigned short* __restrict__ A,
                  const unsigned short* __restrict__ B,
                  const float* __restrict__ bias,
                  float* __restrict__ C,
                  int M, int N, int K, int relu) {
  // dynamic LDS: stage s at byte s*16384; A tile = 8192 B, B tile = 8192 B
  extern __shared__ __align__(16) unsigned short sm16[];

  const int tid  = threadIdx.x;
  const int lane = tid & 31;
  const int wave = tid >> 5;
  const int wm   = wave & 3;   // M quadrant: rows wm*32 .. +31
  const int wn   = wave >> 2;  // N half:    cols wn*64 .. +63
  const int mb   = blockIdx.y * 128;
  const int nb   = blockIdx.x * 128;

  const int h  = lane >> 4;    // half-lane selector (K split)
  const int ml = lane & 15;    // m (or n) within 16

  // LDS byte offset of the dynamic-shared base for async VDST addressing
  const unsigned ldsBase = __builtin_amdgcn_groupstaticsize();

  // each thread owns 16B chunks c and c+256 of the 512-chunk (8 KB) tile:
  // chunk c -> row = c>>2 (0..127), 16B column-chunk cc = c&3
  const int c1 = tid;
  const int c2 = tid + 256;
  const int r1 = c1 >> 2, cc1 = c1 & 3;
  const int r2 = c2 >> 2, cc2 = c2 & 3;

  auto issue_stage = [&](int stage, int k0) {
    const unsigned sb = ldsBase + (unsigned)stage * 16384u;
    // A tile -> LDS bytes [sb, sb+8192)
    {
      unsigned long long ga1 =
          (unsigned long long)(size_t)(A + (size_t)(mb + r1) * K + k0 + cc1 * 8);
      unsigned long long ga2 =
          (unsigned long long)(size_t)(A + (size_t)(mb + r2) * K + k0 + cc2 * 8);
      unsigned l1 = sb + (unsigned)c1 * 16u;
      unsigned l2 = sb + (unsigned)c2 * 16u;
      asm volatile("global_load_async_to_lds_b128 %0, %1, off"
                   :: "v"(l1), "v"(ga1) : "memory");
      asm volatile("global_load_async_to_lds_b128 %0, %1, off"
                   :: "v"(l2), "v"(ga2) : "memory");
    }
    // B tile -> LDS bytes [sb+8192, sb+16384)
    {
      unsigned long long gb1 =
          (unsigned long long)(size_t)(B + (size_t)(nb + r1) * K + k0 + cc1 * 8);
      unsigned long long gb2 =
          (unsigned long long)(size_t)(B + (size_t)(nb + r2) * K + k0 + cc2 * 8);
      unsigned l1 = sb + 8192u + (unsigned)c1 * 16u;
      unsigned l2 = sb + 8192u + (unsigned)c2 * 16u;
      asm volatile("global_load_async_to_lds_b128 %0, %1, off"
                   :: "v"(l1), "v"(gb1) : "memory");
      asm volatile("global_load_async_to_lds_b128 %0, %1, off"
                   :: "v"(l2), "v"(gb2) : "memory");
    }
  };

  v8f acc[2][4] = {};
  union Frag { uint4 q[2]; v16bf v; };

  issue_stage(0, 0);

  const int nk = K / 32;
  for (int i = 0; i < nk; ++i) {
    const int cur = i & 1;
    // my async copies for stage `cur` (4 in flight) must have landed
    asm volatile("s_wait_asynccnt 0x0" ::: "memory");
    // everyone's copies landed + everyone done reading buffer `cur^1`
    __syncthreads();
    if (i + 1 < nk) issue_stage(cur ^ 1, (i + 1) * 32);  // overlaps compute

    const unsigned short* Ac = sm16 + cur * 8192;        // ushort units
    const unsigned short* Bc = sm16 + cur * 8192 + 4096;

    // fragments per ISA 16-bit A/B layout:
    // lane (x + 16h) holds row x, elements e: k = (e&7) + 8h + 16*(e>>3)
    Frag af[2], bf[4];
#pragma unroll
    for (int ii = 0; ii < 2; ++ii) {
      int m = wm * 32 + ii * 16 + ml;
      af[ii].q[0] = *reinterpret_cast<const uint4*>(&Ac[m * 32 + 8 * h]);
      af[ii].q[1] = *reinterpret_cast<const uint4*>(&Ac[m * 32 + 16 + 8 * h]);
    }
#pragma unroll
    for (int j = 0; j < 4; ++j) {
      int n = wn * 64 + j * 16 + ml;
      bf[j].q[0] = *reinterpret_cast<const uint4*>(&Bc[n * 32 + 8 * h]);
      bf[j].q[1] = *reinterpret_cast<const uint4*>(&Bc[n * 32 + 16 + 8 * h]);
    }
#pragma unroll
    for (int ii = 0; ii < 2; ++ii)
#pragma unroll
      for (int j = 0; j < 4; ++j)
        acc[ii][j] = __builtin_amdgcn_wmma_f32_16x16x32_bf16(
            false, af[ii].v, false, bf[j].v, (short)0, acc[ii][j], false, false);
  }

  // epilogue: D layout — lane n = ml, VGPR v -> m = v + 8h
#pragma unroll
  for (int j = 0; j < 4; ++j) {
    int n = nb + wn * 64 + j * 16 + ml;
    float bv = bias ? bias[n] : 0.0f;
#pragma unroll
    for (int ii = 0; ii < 2; ++ii) {
      int mBase = mb + wm * 32 + ii * 16 + 8 * h;
#pragma unroll
      for (int v = 0; v < 8; ++v) {
        float x = acc[ii][j][v] + bv;
        if (relu) x = fmaxf(x, 0.0f);
        C[(size_t)(mBase + v) * N + n] = x;
      }
    }
  }
}

// ---------------- exact top-k via radix-select in big LDS ----------------
// One 256-thread WG per row. Row = 32768 non-negative floats (post-ReLU),
// so uint bit pattern order == value order. 4 passes of 8-bit histograms.
__global__ __launch_bounds__(256)
void topk_rows(const float* __restrict__ acts,
               int* __restrict__ tIdx, float* __restrict__ tVal) {
  extern __shared__ unsigned int sm[];
  unsigned int* sv   = sm;            // 32768 values (128 KB)
  unsigned int* hist = sm + N_FEAT;   // 256 bins
  __shared__ unsigned int s_prefix, s_remaining, s_cntGt, s_cntEq;

  const int row = blockIdx.x;
  const int tid = threadIdx.x;
  const float* rp = acts + (size_t)row * N_FEAT;

  for (int i = tid; i < N_FEAT; i += 256) sv[i] = __float_as_uint(rp[i]);
  if (tid == 0) { s_prefix = 0; s_remaining = TOPK; s_cntGt = 0; s_cntEq = 0; }
  __syncthreads();

  for (int shift = 24; shift >= 0; shift -= 8) {
    if (tid < 256) hist[tid] = 0;
    __syncthreads();
    unsigned int prefix = s_prefix;
    for (int i = tid; i < N_FEAT; i += 256) {
      unsigned int u = sv[i];
      bool match = (shift == 24) || (((u ^ prefix) >> (shift + 8)) == 0);
      if (match) atomicAdd(&hist[(u >> shift) & 255u], 1u);
    }
    __syncthreads();
    if (tid == 0) {
      unsigned int rem = s_remaining, cum = 0;
      int b = 255;
      for (; b > 0; --b) {
        unsigned int c = hist[b];
        if (cum + c >= rem) break;
        cum += c;
      }
      s_prefix = prefix | ((unsigned int)b << shift);
      s_remaining = rem - cum;   // how many to take inside bin b
    }
    __syncthreads();
  }

  const unsigned int thr    = s_prefix;       // exact k-th largest bit pattern
  const unsigned int needEq = s_remaining;    // ties to take at threshold
  const unsigned int eqBase = TOPK - needEq;  // count strictly greater
  int*   oi = tIdx + row * TOPK;
  float* ov = tVal + row * TOPK;

  for (int i = tid; i < N_FEAT; i += 256) {
    unsigned int u = sv[i];
    if (u > thr) {
      unsigned int p = atomicAdd(&s_cntGt, 1u);
      oi[p] = i; ov[p] = __uint_as_float(u);
    } else if (u == thr) {
      unsigned int p = atomicAdd(&s_cntEq, 1u);
      if (p < needEq) { oi[eqBase + p] = i; ov[eqBase + p] = __uint_as_float(u); }
    }
  }
}

// ------------- sparsify acts row + sparse decode into output -------------
// One WG per token: zero 32768-row, scatter top-64, out[t,:] += sum v*W_dec[i,:]
__global__ __launch_bounds__(256)
void sparsify_decode(const int* __restrict__ tIdx, const float* __restrict__ tVal,
                     const float* __restrict__ Wdec,
                     float* __restrict__ outMain, float* __restrict__ acts) {
  __shared__ int   sIdx[TOPK];
  __shared__ float sVal[TOPK];
  const int t = blockIdx.x, tid = threadIdx.x;
  if (tid < TOPK) { sIdx[tid] = tIdx[t * TOPK + tid]; sVal[tid] = tVal[t * TOPK + tid]; }
  __syncthreads();

  float4 z = make_float4(0.f, 0.f, 0.f, 0.f);
  float4* rowv = reinterpret_cast<float4*>(acts + (size_t)t * N_FEAT);
  for (int i = tid; i < N_FEAT / 4; i += 256) rowv[i] = z;
  __syncthreads();
  if (tid < TOPK) acts[(size_t)t * N_FEAT + sIdx[tid]] = sVal[tid];

  float accv[8];
#pragma unroll
  for (int r = 0; r < 8; ++r) accv[r] = 0.f;
  for (int j = 0; j < TOPK; ++j) {
    const float* wr = Wdec + (size_t)sIdx[j] * D_OUT;
    float v = sVal[j];
#pragma unroll
    for (int r = 0; r < 8; ++r)
      accv[r] = fmaf(v, wr[tid + 256 * r], accv[r]);
  }
  float* op = outMain + (size_t)t * D_OUT;
#pragma unroll
  for (int r = 0; r < 8; ++r) op[tid + 256 * r] += accv[r];
}

// --------------------------------- launch ---------------------------------
extern "C" void kernel_launch(void* const* d_in, const int* in_sizes, int n_in,
                              void* d_out, int out_size, void* d_ws, size_t ws_size,
                              hipStream_t stream) {
  const float* x      = (const float*)d_in[0];
  const float* W_enc  = (const float*)d_in[1];
  const float* b_enc  = (const float*)d_in[2];
  const float* W_dec  = (const float*)d_in[3];
  const float* b_dec  = (const float*)d_in[4];
  const float* W_skip = (const float*)d_in[5];
  // d_in[6] is k (==64), compile-time constant here.

  float* out    = (float*)d_out;                      // [TOKENS, D_OUT]
  float* sparse = out + (size_t)TOKENS * D_OUT;       // [TOKENS, N_FEAT]

  char* ws = (char*)d_ws;
  size_t off = 0;
  unsigned short* xb  = (unsigned short*)(ws + off); off += (size_t)TOKENS * D_IN * 2;   // 16 MB
  unsigned short* web = (unsigned short*)(ws + off); off += (size_t)N_FEAT * D_IN * 2;   // 128 MB
  unsigned short* wkb = (unsigned short*)(ws + off); off += (size_t)D_OUT * D_IN * 2;    // 8 MB
  int*   tIdx = (int*)(ws + off);   off += (size_t)TOKENS * TOPK * sizeof(int);          // 1 MB
  float* tVal = (float*)(ws + off); off += (size_t)TOKENS * TOPK * sizeof(float);        // 1 MB

  // 1) bf16 conversions (x + W_enc then fit in 192 MB L2 for the big GEMM)
  cvt_f32_bf16<<<4096, 256, 0, stream>>>(x,      xb,  (TOKENS * D_IN) / 4);
  cvt_f32_bf16<<<8192, 256, 0, stream>>>(W_enc,  web, (N_FEAT * D_IN) / 4);
  cvt_f32_bf16<<<2048, 256, 0, stream>>>(W_skip, wkb, (D_OUT * D_IN) / 4);

  // 2) encode: acts = relu(x @ W_enc.T + b_enc) -> staged in sparse output region
  dim3 gEnc(N_FEAT / 128, TOKENS / 128);
  wmma_gemm_bt<<<gEnc, 256, 32768, stream>>>(xb, web, b_enc, sparse,
                                             TOKENS, N_FEAT, D_IN, 1);

  // 3) skip path: out = x @ W_skip.T + b_dec
  dim3 gSkip(D_OUT / 128, TOKENS / 128);
  wmma_gemm_bt<<<gSkip, 256, 32768, stream>>>(xb, wkb, b_dec, out,
                                              TOKENS, D_OUT, D_IN, 0);

  // 4) exact per-row top-64 (radix select in 129 KB dynamic LDS)
  size_t shbytes = (size_t)(N_FEAT + 256) * sizeof(unsigned int);
  topk_rows<<<TOKENS, 256, shbytes, stream>>>(sparse, tIdx, tVal);

  // 5) zero+scatter sparse_acts rows and accumulate sparse decode into out
  sparsify_decode<<<TOKENS, 256, 0, stream>>>(tIdx, tVal, W_dec, out, sparse);
}
